// loTeNet_12979391169073
// MI455X (gfx1250) — matrix operand order repaired and checked by reading, not compiled, not run on hardware
//
#include <hip/hip_runtime.h>

// ---------------------------------------------------------------------------
// loTeNet forward for MI455X (gfx1250, wave32, WMMA + Tensor Data Mover).
//
// Stage k (k=1..3): one workgroup per expert p.
//   - load the 4 site features (float4, contiguous) -> LDS as f16 [x ; 1-x]
//   - per site: GEMM  M[b, u*20+v] = sum_f feat[b,f] * cores[p,site,u,f,v]
//     via v_wmma_f32_16x16x32_f16, N tiled in 5 chunks of 80 cols (4 u-rows)
//   - weight chunks are DMA'd global->LDS with TENSOR_LOAD_TO_LDS (double
//     buffered, overlapped with WMMA of the previous chunk; TENSORcnt sync)
//   - fold each chunk into the running left-vector  left' = left * M
//   - label contraction + fused training-mode BatchNorm (channel == p)
// Final stage: 16-site chain per batch element, plain VALU (tiny).
// ---------------------------------------------------------------------------

typedef _Float16 f16;
typedef _Float16 v16h __attribute__((ext_vector_type(16)));
typedef float    v8f  __attribute__((ext_vector_type(8)));
typedef unsigned int u32x4 __attribute__((ext_vector_type(4)));
typedef int          i32x4 __attribute__((ext_vector_type(4)));
typedef int          i32x8 __attribute__((ext_vector_type(8)));

union AFrag { v16h v; f16   h[16]; };
union CFrag { v8f  v; float f[8];  };

#define BN_EPSF     1e-5f
#define INV_SQRT_D  0.22360679774997896f   // 20^-0.5

#if defined(__has_builtin)
#  if __has_builtin(__builtin_amdgcn_tensor_load_to_lds) && \
      __has_builtin(__builtin_amdgcn_s_wait_tensorcnt)
#    define USE_TDM 1
#  endif
#endif

#ifdef USE_TDM
typedef __attribute__((address_space(3))) char* lds_as3_t;

__device__ __forceinline__ unsigned lds_byte_off(const void* p) {
    return (unsigned)(unsigned long long)(lds_as3_t)(char*)p;
}

// 1-D contiguous TDM copy: nElems f32 from gsrc -> LDS at lds_addr (bytes).
// Descriptor packing per CDNA5 ISA section 8.3/8.4 (D# group 0 + group 1).
__device__ __forceinline__ void tdm_load_1d(const float* gsrc, unsigned lds_addr,
                                            unsigned nElems)
{
    unsigned long long ga = (unsigned long long)gsrc;
    u32x4 g0;
    g0[0] = 1u;                                   // count=1, user descriptor
    g0[1] = lds_addr;                             // lds_addr[31:0]
    g0[2] = (unsigned)(ga & 0xFFFFFFFFu);         // global_addr[31:0]
    g0[3] = (unsigned)((ga >> 32) & 0x01FFFFFFu)  // global_addr[56:32]
          | 0x80000000u;                          // type=2 ("image")
    i32x8 g1;
    g1[0] = 0x00020000;                           // data_size=2 (4 bytes)
    g1[1] = (int)((nElems & 0xFFFFu) << 16);      // tensor_dim0[15:0]
    g1[2] = (int)(((nElems >> 16) & 0xFFFFu)      // tensor_dim0[31:16]
          | (1u << 16));                          // tensor_dim1 = 1
    g1[3] = (int)((nElems & 0xFFFFu) << 16);      // tile_dim0 = nElems
    g1[4] = 1;                                    // tile_dim1 = 1, tile_dim2 = 0
    g1[5] = (int)nElems;                          // tensor_dim0_stride[31:0]
    g1[6] = 0;
    g1[7] = 0;
    i32x4 z4 = {0, 0, 0, 0};
#if __clang_major__ >= 23
    i32x8 z8 = {0, 0, 0, 0, 0, 0, 0, 0};
    __builtin_amdgcn_tensor_load_to_lds(g0, g1, z4, z4, z8, 0);
#else
    __builtin_amdgcn_tensor_load_to_lds(g0, g1, z4, z4, 0);
#endif
}
#endif  // USE_TDM

template<int S> struct Cfg;
template<> struct Cfg<1> { static constexpr int P = 256, C = 48, F = 96, KP = 96, KSTEPS = 3; };
template<> struct Cfg<2> { static constexpr int P = 64,  C = 20, F = 40, KP = 64, KSTEPS = 2; };
template<> struct Cfg<3> { static constexpr int P = 16,  C = 20, F = 40, KP = 64, KSTEPS = 2; };

template<int STAGE>
__global__ __launch_bounds__(256)
void mps_stage(const float* __restrict__ xin,
               const float* __restrict__ cores,
               const float* __restrict__ label,
               const float* __restrict__ gamma,
               const float* __restrict__ beta,
               float* __restrict__ yout)
{
    constexpr int P      = Cfg<STAGE>::P;
    constexpr int C      = Cfg<STAGE>::C;
    constexpr int F      = Cfg<STAGE>::F;     // 2*C
    constexpr int KP     = Cfg<STAGE>::KP;    // K padded to multiple of 32
    constexpr int KSTEPS = Cfg<STAGE>::KSTEPS;
    constexpr int FPAD   = KP + 8;            // row pad (16B) vs LDS bank clash
    constexpr int CHUNK  = 4 * F * 20;        // floats per (site,u-chunk) weight block

    __shared__ f16   featAll[4][128][FPAD];   // f16 features, all 4 sites
    __shared__ f16   Bt[80][KP];              // weight chunk, n-major (b128 frag loads)
    __shared__ float Mc[128][80];             // GEMM output chunk
    __shared__ float leftL[128][20];
    __shared__ float lnew[128][20];
    __shared__ float Lp[400];                 // label . boundary-vector
    __shared__ float yb[128][20];
    __shared__ float red1[256], red2[256];
    __shared__ float bc[2];
#ifdef USE_TDM
    __shared__ __align__(16) float rawF[2][CHUNK];  // TDM double buffer (raw f32)
#endif

    const int p    = blockIdx.x;
    const int tid  = threadIdx.x;
    const int wave = tid >> 5;
    const int lane = tid & 31;
    const int l16  = lane & 15;
    const int half = lane >> 4;

    const long coresBase = (long)p * 20 * CHUNK;  // chunks contiguous across sites

#ifdef USE_TDM
    // Kick off the DMA of the first weight chunk immediately.
    if (wave == 0)
        tdm_load_1d(cores + coresBase, lds_byte_off(&rawF[0][0]), CHUNK);
#endif

    // Uniform (SGPR) patch coordinates for this expert.
    int h, w0;
    if constexpr (STAGE == 1) {
        int idx0 = p * 4;
        h  = ((idx0 >> 9) & 1) * 16 + ((idx0 >> 4) & 15);
        w0 = ((idx0 >> 8) & 1) * 16 + (idx0 & 15);
    } else if constexpr (STAGE == 2) {
        int idx0 = p * 4;
        h  = ((idx0 >> 7) & 1) * 8 + ((idx0 >> 3) & 7);
        w0 = ((idx0 >> 6) & 1) * 8 + (idx0 & 7);
    } else {
        int idx0 = p * 4;
        h  = ((idx0 >> 5) & 1) * 4 + ((idx0 >> 2) & 3);
        w0 = ((idx0 >> 4) & 1) * 4 + (idx0 & 3);
    }

    // Zero the K padding once (never overwritten afterwards).
    if constexpr (KP > F) {
        constexpr int PADW = KP - F;
        for (int e = tid; e < 128 * PADW; e += 256) {
            int b = e / PADW, k = F + e % PADW;
            featAll[0][b][k] = (f16)0.f; featAll[1][b][k] = (f16)0.f;
            featAll[2][b][k] = (f16)0.f; featAll[3][b][k] = (f16)0.f;
        }
        for (int e = tid; e < 80 * PADW; e += 256)
            Bt[e / PADW][F + e % PADW] = (f16)0.f;
    }

    // Feature map: the 4 site values of (b,c) are 4 contiguous floats.
    for (int e = tid; e < 128 * C; e += 256) {
        int b = e / C, c = e % C;
        long base;
        if constexpr (STAGE == 1)      base = ((long)(b * 48 + c) * 32 + h) * 32 + w0;
        else if constexpr (STAGE == 2) base = (long)b * 5120 + c * 256 + h * 16 + w0;
        else                           base = (long)b * 1280 + c * 64  + h * 8  + w0;
        const float4 xv = *reinterpret_cast<const float4*>(xin + base);
        float vals[4] = {xv.x, xv.y, xv.z, xv.w};
#pragma unroll
        for (int n = 0; n < 4; ++n) {
            featAll[n][b][c]     = (f16)vals[n];
            featAll[n][b][C + c] = (f16)(1.f - vals[n]);
        }
    }

    // Lp[u][o] = sum_v label[p,u,o,v] * v_boundary
    for (int e = tid; e < 400; e += 256) {
        int u = e / 20, o = e % 20;
        const float* lp = label + (((long)p * 20 + u) * 20 + o) * 20;
        float s = 0.f;
#pragma unroll
        for (int v = 0; v < 20; ++v) s += lp[v];
        Lp[e] = s * INV_SQRT_D;
    }

    for (int e = tid; e < 2560; e += 256) (&leftL[0][0])[e] = INV_SQRT_D;
    __syncthreads();

    // Flattened (site x u-chunk) pipeline: 4 sites x 5 chunks.
    AFrag a[KSTEPS];
    for (int it = 0; it < 20; ++it) {
        const int site = it / 5;
        const int cu   = it % 5;

        if (cu == 0) {
            // New site: reset chain accumulator, hoist A fragments
            // (16-bit WMMA A layout: lanes 0-15 row l16 K{0..7}+{16..23},
            //  lanes 16-31 K{8..15}+{24..31}).
            for (int e = tid; e < 2560; e += 256) (&lnew[0][0])[e] = 0.f;
            const f16* rp = &featAll[site][wave * 16 + l16][0];
#pragma unroll
            for (int ks = 0; ks < KSTEPS; ++ks) {
                int b1 = ks * 32 + half * 8, b2 = b1 + 16;
#pragma unroll
                for (int i = 0; i < 8; ++i) {
                    a[ks].h[i]     = rp[b1 + i];
                    a[ks].h[8 + i] = rp[b2 + i];
                }
            }
        }

#ifdef USE_TDM
        const int buf = it & 1;
        if (wave == 0) {
            if (it < 19) {
                // DMA the next chunk into the other buffer, then wait until
                // only that newest op is outstanding (TDM retires in order).
                tdm_load_1d(cores + coresBase + (long)(it + 1) * CHUNK,
                            lds_byte_off(&rawF[buf ^ 1][0]), CHUNK);
                __builtin_amdgcn_s_wait_tensorcnt(1);
            } else {
                __builtin_amdgcn_s_wait_tensorcnt(0);
            }
        }
        __syncthreads();  // rawF[buf] visible to all waves

        // Convert raw f32 chunk -> f16 Bt in WMMA B order (n-major).
        for (int e = tid; e < CHUNK; e += 256) {
            int u = e / (F * 20), rem = e % (F * 20);
            int f = rem / 20, v = rem % 20;
            Bt[u * 20 + v][f] = (f16)rawF[buf][e];
        }
#else
        const long cbase = coresBase + (long)it * CHUNK;
        for (int e = tid; e < CHUNK; e += 256) {
            int u = e / (F * 20), rem = e % (F * 20);
            int f = rem / 20, v = rem % 20;
            Bt[u * 20 + v][f] = (f16)cores[cbase + e];
        }
        if (it < 19 && tid * 32 < CHUNK)
            __builtin_prefetch(cores + cbase + CHUNK + tid * 32, 0, 1);
#endif
        __syncthreads();

        // 8 waves x (1 btile x 5 ntiles): M[b, n] += A x B over KSTEPS.
#pragma unroll
        for (int nt = 0; nt < 5; ++nt) {
            CFrag acc;
#pragma unroll
            for (int i = 0; i < 8; ++i) acc.f[i] = 0.f;
            const f16* bp = &Bt[nt * 16 + l16][half * 8];
#pragma unroll
            for (int ks = 0; ks < KSTEPS; ++ks) {
                AFrag bf;
#pragma unroll
                for (int i = 0; i < 16; ++i) bf.h[i] = bp[ks * 32 + i + (i >> 3) * 8];
                acc.v = __builtin_amdgcn_wmma_f32_16x16x32_f16(
                    false, a[ks].v, false, bf.v, (short)0, acc.v, false, false);
            }
            const int row0 = wave * 16 + half * 8;
            const int col  = nt * 16 + l16;
#pragma unroll
            for (int r = 0; r < 8; ++r) Mc[row0 + r][col] = acc.f[r];
        }
        __syncthreads();

        // Fold this chunk (u in [4cu, 4cu+4)) into left' = left * M.
        {
            const int b  = tid >> 1;
            const int vh = (tid & 1) * 10;
            const int u0 = cu * 4;
#pragma unroll
            for (int v = 0; v < 10; ++v) {
                float s = 0.f;
#pragma unroll
                for (int du = 0; du < 4; ++du)
                    s += leftL[b][u0 + du] * Mc[b][du * 20 + vh + v];
                lnew[b][vh + v] += s;
            }
        }
        __syncthreads();

        if (cu == 4)
            for (int e = tid; e < 2560; e += 256)
                (&leftL[0][0])[e] = (&lnew[0][0])[e];
        __syncthreads();
    }

    // y[b,o] = left . Lp, then fused training-mode BN over (b,o) for channel p.
    const int b  = tid >> 1;
    const int oh = (tid & 1) * 10;
    float s1 = 0.f, s2 = 0.f;
#pragma unroll
    for (int o = 0; o < 10; ++o) {
        float s = 0.f;
#pragma unroll
        for (int u = 0; u < 20; ++u) s += leftL[b][u] * Lp[u * 20 + oh + o];
        yb[b][oh + o] = s; s1 += s; s2 += s * s;
    }
    red1[tid] = s1; red2[tid] = s2;
    __syncthreads();
    for (int off = 128; off > 0; off >>= 1) {
        if (tid < off) { red1[tid] += red1[tid + off]; red2[tid] += red2[tid + off]; }
        __syncthreads();
    }
    if (tid == 0) {
        float mean = red1[0] * (1.f / 2560.f);
        float var  = red2[0] * (1.f / 2560.f) - mean * mean;
        float sc   = rsqrtf(var + BN_EPSF) * gamma[p];
        bc[0] = sc; bc[1] = beta[p] - mean * sc;
    }
    __syncthreads();
    const float sc = bc[0], sh = bc[1];
#pragma unroll
    for (int o = 0; o < 10; ++o)
        yout[((long)b * P + p) * 20 + oh + o] = yb[b][oh + o] * sc + sh;
}

// Final MPS over 16 expert outputs -> [128, 10]. Tiny; VALU only.
__global__ __launch_bounds__(256)
void mps_final(const float* __restrict__ y3,
               const float* __restrict__ coresF,
               const float* __restrict__ labelF,
               float* __restrict__ out)
{
    __shared__ float yrow[320];
    __shared__ float M[400];
    __shared__ float left[20], nl[20];
    __shared__ float LF[200];
    const int b = blockIdx.x, tid = threadIdx.x;

    for (int e = tid; e < 320; e += 256) yrow[e] = y3[(long)b * 320 + e];
    for (int e = tid; e < 200; e += 256) {
        int u = e / 10, o = e % 10;
        const float* lp = labelF + (u * 10 + o) * 20;
        float s = 0.f;
#pragma unroll
        for (int v = 0; v < 20; ++v) s += lp[v];
        LF[e] = s * INV_SQRT_D;
    }
    if (tid < 20) left[tid] = INV_SQRT_D;
    __syncthreads();

    for (int s = 0; s < 16; ++s) {
        for (int e = tid; e < 400; e += 256) {
            int u = e / 20, v = e % 20;
            const float* cp = coresF + ((long)(s * 20 + u) * 40) * 20 + v;
            float acc = 0.f;
#pragma unroll
            for (int f = 0; f < 20; ++f) {
                float xv = yrow[s * 20 + f];
                acc += xv * cp[f * 20] + (1.f - xv) * cp[(20 + f) * 20];
            }
            M[e] = acc;
        }
        __syncthreads();
        if (tid < 20) {
            float a = 0.f;
#pragma unroll
            for (int u = 0; u < 20; ++u) a += left[u] * M[u * 20 + tid];
            nl[tid] = a;
        }
        __syncthreads();
        if (tid < 20) left[tid] = nl[tid];
        __syncthreads();
    }
    if (tid < 10) {
        float a = 0.f;
#pragma unroll
        for (int u = 0; u < 20; ++u) a += left[u] * LF[u * 10 + tid];
        out[b * 10 + tid] = a;
    }
}

extern "C" void kernel_launch(void* const* d_in, const int* in_sizes, int n_in,
                              void* d_out, int out_size, void* d_ws, size_t ws_size,
                              hipStream_t stream)
{
    (void)in_sizes; (void)n_in; (void)out_size; (void)ws_size;
    const float* x      = (const float*)d_in[0];
    const float* cores1 = (const float*)d_in[1];
    const float* label1 = (const float*)d_in[2];
    const float* g1     = (const float*)d_in[3];
    const float* b1     = (const float*)d_in[4];
    const float* cores2 = (const float*)d_in[5];
    const float* label2 = (const float*)d_in[6];
    const float* g2     = (const float*)d_in[7];
    const float* b2     = (const float*)d_in[8];
    const float* cores3 = (const float*)d_in[9];
    const float* label3 = (const float*)d_in[10];
    const float* g3     = (const float*)d_in[11];
    const float* b3     = (const float*)d_in[12];
    const float* coresF = (const float*)d_in[13];
    const float* labelF = (const float*)d_in[14];

    float* y1 = (float*)d_ws;                    // [128][256][20]
    float* y2 = y1 + 128 * 256 * 20;             // [128][64][20]
    float* y3 = y2 + 128 * 64 * 20;              // [128][16][20]

    mps_stage<1><<<256, 256, 0, stream>>>(x,  cores1, label1, g1, b1, y1);
    mps_stage<2><<<64,  256, 0, stream>>>(y1, cores2, label2, g2, b2, y2);
    mps_stage<3><<<16,  256, 0, stream>>>(y2, cores3, label3, g3, b3, y3);
    mps_final<<<128, 256, 0, stream>>>(y3, coresF, labelF, (float*)d_out);
}
